// adaptIndexSplit_dotsim_81312320848589
// MI455X (gfx1250) — matrix operand order repaired and checked by reading, not compiled, not run on hardware
//
#include <hip/hip_runtime.h>
#include <math.h>

typedef __attribute__((ext_vector_type(16))) _Float16 v16h;
typedef __attribute__((ext_vector_type(8)))  _Float16 v8h;
typedef __attribute__((ext_vector_type(8)))  float    v8f;
typedef __attribute__((ext_vector_type(4)))  float    v4f;

#define BB   32
#define CC   512
#define TT   16
#define HWW  196
#define KDIM 25088   /* 49*512 */
#define MDIM 512     /* B*T   */
#define NDIM 512

// ---------------- K1: 2x2 avg pool + fp16 pack into X[m][k], k=(ph*7+pw)*512+c
__global__ __launch_bounds__(256) void pool_pack_kernel(const float* __restrict__ x,
                                                        _Float16* __restrict__ X) {
  long id = (long)blockIdx.x * blockDim.x + threadIdx.x;   // MDIM*KDIM threads
  int c    = (int)(id & 511);
  long r   = id >> 9;
  int pp   = (int)(r % 49);
  int m    = (int)(r / 49);
  int ph   = pp / 7, pw = pp % 7;
  int b    = m >> 4, t = m & 15;
  const float* base = x + (((long)(b * CC + c)) * TT + t) * HWW;
  int h0 = 2 * ph, w0 = 2 * pw;
  float s = base[h0 * 14 + w0]       + base[h0 * 14 + w0 + 1]
          + base[(h0 + 1) * 14 + w0] + base[(h0 + 1) * 14 + w0 + 1];
  X[id] = (_Float16)(0.25f * s);
}

// ---------------- K2: W1 (25088x512 f32) -> W1T (512x25088 f16), LDS-tiled 32x32
// Coalesced on both sides; padded LDS row (33) avoids bank conflicts.
__global__ __launch_bounds__(256) void w1_transpose_kernel(const float* __restrict__ W1,
                                                           _Float16* __restrict__ W1T) {
  __shared__ float tile[32][33];
  int tk = blockIdx.x % (KDIM / 32);   // 784 K tiles
  int tn = blockIdx.x / (KDIM / 32);   // 16 N tiles
  int K0 = tk * 32, N0 = tn * 32;
  int tid = threadIdx.x;
#pragma unroll
  for (int e = tid; e < 1024; e += 256) {
    int i = e >> 5, j = e & 31;        // i: k-in-tile, j: n-in-tile (coalesced along n)
    tile[i][j] = W1[(long)(K0 + i) * NDIM + N0 + j];
  }
  __syncthreads();
#pragma unroll
  for (int e = tid; e < 1024; e += 256) {
    int j = e >> 5, i = e & 31;        // coalesced along k in the output
    W1T[(long)(N0 + j) * KDIM + K0 + i] = (_Float16)tile[i][j];
  }
}

// ---------------- K3: H = gelu(X @ W1 + b1) via v_wmma_f32_16x16x32_f16
// 2x2 register blocking: one wave32 computes a 32x32 macro-tile (4 WMMA tiles).
// 16x16 macro-tile grid -> 256 waves -> 32 blocks x 8 waves.
__global__ __launch_bounds__(256) void gemm1_wmma_kernel(const _Float16* __restrict__ X,
                                                         const _Float16* __restrict__ W1T,
                                                         const float* __restrict__ b1,
                                                         float* __restrict__ H) {
  int lane = threadIdx.x & 31;
  int wave = threadIdx.x >> 5;
  int gw   = blockIdx.x * 8 + wave;          // 0..255
  int tm   = gw >> 4;                        // 0..15 (32-row macro tile)
  int tn   = gw & 15;                        // 0..15 (32-col macro tile)
  int r16  = lane & 15;
  int hi   = lane >> 4;                      // 0 or 1

  // A fragment (16x32 f16): lane<16 holds row r16, K 0..15;
  // lane>=16 holds row r16, K 16..31.
  const _Float16* Arow0 = X + (long)(tm * 32 + r16) * KDIM;
  const _Float16* Arow1 = Arow0 + (long)16 * KDIM;
  // B fragment (32x16 f16): lane = column n, K 0..15 (lanes 0-15) / 16..31
  // (lanes 16-31), contiguous along K because W1T is N-major.
  const _Float16* Brow0 = W1T + (long)(tn * 32 + r16) * KDIM;
  const _Float16* Brow1 = Brow0 + (long)16 * KDIM;

  v8f acc00 = {}, acc01 = {}, acc10 = {}, acc11 = {};
  for (int k0 = 0; k0 < KDIM; k0 += 32) {
    v16h a0 = *(const v16h*)(Arow0 + k0 + hi * 16);
    v16h a1 = *(const v16h*)(Arow1 + k0 + hi * 16);
    v16h bf0 = *(const v16h*)(Brow0 + k0 + hi * 16);
    v16h bf1 = *(const v16h*)(Brow1 + k0 + hi * 16);
    __builtin_prefetch(Arow0 + k0 + 256, 0, 3);
    __builtin_prefetch(Arow1 + k0 + 256, 0, 3);
    __builtin_prefetch(Brow0 + k0 + 256, 0, 3);
    __builtin_prefetch(Brow1 + k0 + 256, 0, 3);
    acc00 = __builtin_amdgcn_wmma_f32_16x16x32_f16(false, a0, false, bf0, (short)0, acc00, false, false);
    acc01 = __builtin_amdgcn_wmma_f32_16x16x32_f16(false, a0, false, bf1, (short)0, acc01, false, false);
    acc10 = __builtin_amdgcn_wmma_f32_16x16x32_f16(false, a1, false, bf0, (short)0, acc10, false, false);
    acc11 = __builtin_amdgcn_wmma_f32_16x16x32_f16(false, a1, false, bf1, (short)0, acc11, false, false);
  }

  // D layout: VGPR r -> M = r (lanes 0-15) / r+8 (lanes 16-31); N = lane&15.
  int n0 = tn * 32 + r16;
  int n1 = n0 + 16;
  float bias0 = b1[n0];
  float bias1 = b1[n1];
#pragma unroll
  for (int r = 0; r < 8; ++r) {
    int m0 = tm * 32 + r + hi * 8;
    int m1 = m0 + 16;
    float z, g;
    z = acc00[r] + bias0;
    g = 0.5f * z * (1.0f + erff(z * 0.70710678118654752f));
    H[(long)m0 * NDIM + n0] = g;
    z = acc01[r] + bias1;
    g = 0.5f * z * (1.0f + erff(z * 0.70710678118654752f));
    H[(long)m0 * NDIM + n1] = g;
    z = acc10[r] + bias0;
    g = 0.5f * z * (1.0f + erff(z * 0.70710678118654752f));
    H[(long)m1 * NDIM + n0] = g;
    z = acc11[r] + bias1;
    g = 0.5f * z * (1.0f + erff(z * 0.70710678118654752f));
    H[(long)m1 * NDIM + n1] = g;
  }
}

// ---------------- K4: feat = H@W2+b2 (per batch), attn softmax, offsets
__global__ __launch_bounds__(256) void attn_offset_kernel(const float* __restrict__ H,
                                                          const float* __restrict__ W2,
                                                          const float* __restrict__ b2,
                                                          const float* __restrict__ Wa,
                                                          const float* __restrict__ Wd,
                                                          float* __restrict__ offA,
                                                          float* __restrict__ offD) {
  __shared__ float feat[16][64];
  __shared__ float flat[256];
  __shared__ float logit[16][16];
  int b = blockIdx.x;
  int tid = threadIdx.x;

  for (int e = tid; e < 16 * 64; e += 256) {
    int r = e >> 6, n = e & 63;
    const float* hrow = H + (long)(b * 16 + r) * NDIM;
    float s = b2[n];
    for (int k = 0; k < NDIM; ++k) s += hrow[k] * W2[k * 64 + n];
    feat[r][n] = s;
  }
  __syncthreads();
  {
    int r = tid >> 4, s_ = tid & 15;
    float d = 0.f;
    for (int k = 0; k < 64; ++k) d += feat[r][k] * feat[s_][k];
    logit[r][s_] = d * (1.0f / 512.0f);
  }
  __syncthreads();
  if (tid < 16) {
    float mx = logit[tid][0];
    for (int s_ = 1; s_ < 16; ++s_) mx = fmaxf(mx, logit[tid][s_]);
    float e[16], sum = 0.f;
    for (int s_ = 0; s_ < 16; ++s_) { e[s_] = __expf(logit[tid][s_] - mx); sum += e[s_]; }
    float inv = 1.0f / sum;
    for (int s_ = 0; s_ < 16; ++s_) flat[tid * 16 + s_] = e[s_] * inv;
  }
  __syncthreads();
  if (tid < 16) {
    int j = tid & 7;
    const float* Wm = (tid < 8) ? Wa : Wd;
    float s = 0.f;
    for (int i = 0; i < 256; ++i) s += flat[i] * Wm[i * 8 + j];
    float v = tanhf(s) * 2.0f;
    if (tid < 8) offA[b * 8 + j] = v; else offD[b * 8 + j] = v;
  }
}

// ---------------- K5: temporal bilinear gather, float4-vectorized (HW=196=49*4,
// so every t-slice offset is 16B aligned). b128 loads/stores for HBM rate.
__global__ __launch_bounds__(256) void sample_kernel(const float* __restrict__ x,
                                                     const float* __restrict__ offA,
                                                     const float* __restrict__ offD,
                                                     float* __restrict__ outA,
                                                     float* __restrict__ outD) {
  long id = (long)blockIdx.x * blockDim.x + threadIdx.x;   // B*C*(T/2)*49 float4 threads
  int hw4 = (int)(id % 49);
  long r = id / 49;
  int to = (int)(r & 7);
  long r2 = r >> 3;
  int c = (int)(r2 & 511);
  int b = (int)(r2 >> 9);
  const float* base = x + ((long)(b * CC + c) * TT) * HWW + hw4 * 4;
  float oa = offA[b * 8 + to];
  float od = offD[b * 8 + to];
  {
    float idx = 2.0f * (float)to + oa;
    float p = idx * (16.0f / 15.0f) - 0.5f;
    float f = floorf(p);
    float w = p - f;
    int i0 = (int)f;
    int i0c = min(max(i0, 0), 15);
    int i1c = min(max(i0 + 1, 0), 15);
    v4f g0 = *(const v4f*)(base + (long)i0c * HWW);
    v4f g1 = *(const v4f*)(base + (long)i1c * HWW);
    *(v4f*)(outA + id * 4) = (1.0f - w) * g0 + w * g1;
  }
  {
    float idx = 2.0f * (float)to + 1.0f + od;
    float p = idx * (16.0f / 15.0f) - 0.5f;
    float f = floorf(p);
    float w = p - f;
    int i0 = (int)f;
    int i0c = min(max(i0, 0), 15);
    int i1c = min(max(i0 + 1, 0), 15);
    v4f g0 = *(const v4f*)(base + (long)i0c * HWW);
    v4f g1 = *(const v4f*)(base + (long)i1c * HWW);
    *(v4f*)(outD + id * 4) = (1.0f - w) * g0 + w * g1;
  }
}

extern "C" void kernel_launch(void* const* d_in, const int* in_sizes, int n_in,
                              void* d_out, int out_size, void* d_ws, size_t ws_size,
                              hipStream_t stream) {
  (void)in_sizes; (void)n_in; (void)out_size; (void)ws_size;
  const float* x  = (const float*)d_in[0];
  const float* W1 = (const float*)d_in[1];
  const float* b1 = (const float*)d_in[2];
  const float* W2 = (const float*)d_in[3];
  const float* b2 = (const float*)d_in[4];
  const float* Wa = (const float*)d_in[5];
  const float* Wd = (const float*)d_in[6];

  char* ws = (char*)d_ws;
  _Float16* X    = (_Float16*)(ws);                         // 512*25088*2  = 25,690,112 B
  _Float16* W1T  = (_Float16*)(ws + 25690112L);             // 512*25088*2  = 25,690,112 B
  float*    H    = (float*)   (ws + 51380224L);             // 512*512*4    =  1,048,576 B
  float*    offA = (float*)   (ws + 52428800L);             // 32*8*4
  float*    offD = (float*)   (ws + 52429824L);             // 32*8*4

  float* outA = (float*)d_out;
  float* outD = outA + 25690112L;                           // B*C*(T/2)*HW

  const long nPack = (long)MDIM * KDIM;                     // 12,845,056
  pool_pack_kernel   <<<(int)(nPack / 256), 256, 0, stream>>>(x, X);
  w1_transpose_kernel<<<(KDIM / 32) * (NDIM / 32), 256, 0, stream>>>(W1, W1T);
  gemm1_wmma_kernel  <<<32, 256, 0, stream>>>(X, W1T, b1, H);
  attn_offset_kernel <<<BB, 256, 0, stream>>>(H, W2, b2, Wa, Wd, offA, offD);
  const long nVec4 = 25690112L / 4;                         // float4 elements per output
  sample_kernel      <<<(int)(nVec4 / 256), 256, 0, stream>>>(x, offA, offD, outA, outD);
}
